// MultiHeadSelfAttention_49658411876939
// MI455X (gfx1250) — matrix operand order repaired and checked by reading, not compiled, not run on hardware
//
#include <hip/hip_runtime.h>
#include <hip/hip_bf16.h>

// ---------------------------------------------------------------------------
// Multi-head self-attention, fused flash-style, bf16 WMMA (CDNA5 / gfx1250).
//   B=2, S=1728, E=256, H=8, P=32.
// ~8 GFLOP total; all intermediates L2-resident; scores never touch HBM.
// GEMMs use 16x64-per-wave tiles: 1 A-tile load feeds 4 chained WMMAs.
// ---------------------------------------------------------------------------

typedef __attribute__((ext_vector_type(16))) __bf16 v16bf;
typedef __attribute__((ext_vector_type(8)))  __bf16 v8bf;
typedef __attribute__((ext_vector_type(8)))  float  v8f;

static constexpr int E_  = 256;
static constexpr int S_  = 1728;
static constexpr int B_  = 2;
static constexpr int H_  = 8;
static constexpr int P_  = 32;
static constexpr int BS_ = B_ * S_;   // 3456

// ---- helpers --------------------------------------------------------------

__device__ __forceinline__ v16bf ldpair(const __bf16* p0, const __bf16* p1) {
  // assemble a 16-element bf16 operand from two 16-byte chunks
  v8bf lo = *reinterpret_cast<const v8bf*>(p0);
  v8bf hi = *reinterpret_cast<const v8bf*>(p1);
  v16bf r;
#pragma unroll
  for (int i = 0; i < 8; ++i) { r[i] = lo[i]; r[i + 8] = hi[i]; }
  return r;
}

__device__ __forceinline__ v16bf ld16(const __bf16* p) {
  // 16 contiguous bf16 (B-matrix per-lane column)
  return ldpair(p, p + 8);
}

__device__ __forceinline__ v8f wmma_bf16(v16bf a, v16bf b, v8f c) {
  // D = A(16x32) * B(32x16) + C, f32 accumulate
  return __builtin_amdgcn_wmma_f32_16x16x32_bf16(
      /*neg_a=*/false, a, /*neg_b=*/false, b,
      /*c_mod=*/(short)0, c, /*reuse_a=*/false, /*reuse_b=*/false);
}

__device__ __forceinline__ float half16_max(float v) {
#pragma unroll
  for (int m = 1; m < 16; m <<= 1) v = fmaxf(v, __shfl_xor(v, m, 32));
  return v;
}
__device__ __forceinline__ float half16_sum(float v) {
#pragma unroll
  for (int m = 1; m < 16; m <<= 1) v += __shfl_xor(v, m, 32);
  return v;
}

// ---- conversion kernels ---------------------------------------------------

__global__ void cvt_f32_bf16(const float* __restrict__ x,
                             __bf16* __restrict__ xb, int n) {
  int i = blockIdx.x * blockDim.x + threadIdx.x;
  if (i < n) xb[i] = (__bf16)x[i];
}

// Wt[n][k] = W[k][n]  (so B-matrix operand loads are contiguous per lane)
__global__ void cvt_wT_bf16(const float* __restrict__ w,
                            __bf16* __restrict__ wt) {
  int i = blockIdx.x * blockDim.x + threadIdx.x;
  if (i >= E_ * E_) return;
  int n = i / E_, k = i % E_;
  wt[i] = (__bf16)w[k * E_ + n];
}

// ---- QKV projection: Y = Xb @ W + bias, head-split scatter ----------------
// 16 rows x 64 cols per wave: one A-tile load feeds 4 WMMAs per k-step.
// which: 0 -> Q [B,H,S,P], 1 -> K [B,H,S,P], 2 -> V transposed [B,H,P,S]

__global__ __launch_bounds__(32)
void proj_kernel(const __bf16* __restrict__ Xb, const __bf16* __restrict__ Wt,
                 const float* __restrict__ bias,
                 __bf16* __restrict__ Qb, __bf16* __restrict__ Kb,
                 __bf16* __restrict__ Vtb, int which) {
  const int lane = threadIdx.x;
  const bool lo  = lane < 16;
  const int nl   = lane & 15;
  const int mt   = blockIdx.x;          // row tile over BS (16 rows)
  const int nt0  = blockIdx.y * 64;     // col tile over E  (64 cols)
  const int arow = mt * 16 + nl;

  const __bf16* ap = Xb + (size_t)arow * E_;
  const __bf16* bp0 = Wt + (size_t)(nt0 + nl) * E_;
  const __bf16* bp1 = bp0 + (size_t)16 * E_;
  const __bf16* bp2 = bp0 + (size_t)32 * E_;
  const __bf16* bp3 = bp0 + (size_t)48 * E_;

  v8f acc0 = {}, acc1 = {}, acc2 = {}, acc3 = {};
#pragma unroll
  for (int kk = 0; kk < E_; kk += 32) {
    v16bf a = ld16(ap + kk + (lo ? 0 : 16));
    int ko = kk + (lo ? 0 : 16);
    acc0 = wmma_bf16(a, ld16(bp0 + ko), acc0);
    acc1 = wmma_bf16(a, ld16(bp1 + ko), acc1);
    acc2 = wmma_bf16(a, ld16(bp2 + ko), acc2);
    acc3 = wmma_bf16(a, ld16(bp3 + ko), acc3);
  }

#pragma unroll
  for (int j = 0; j < 4; ++j) {
    const v8f& acc = (j == 0) ? acc0 : (j == 1) ? acc1 : (j == 2) ? acc2 : acc3;
    const int bcol = nt0 + j * 16 + nl;
    const float bb = bias[bcol];
    const int hh = bcol >> 5, pp = bcol & 31;
#pragma unroll
    for (int r = 0; r < 8; ++r) {
      int grow = mt * 16 + r + (lo ? 0 : 8);     // global row in [0, BS)
      int bidx = grow / S_, srow = grow % S_;
      __bf16 v = (__bf16)(acc[r] + bb);
      if (which == 0)
        Qb[(((size_t)bidx * H_ + hh) * S_ + srow) * P_ + pp] = v;
      else if (which == 1)
        Kb[(((size_t)bidx * H_ + hh) * S_ + srow) * P_ + pp] = v;
      else
        Vtb[(((size_t)bidx * H_ + hh) * P_ + pp) * S_ + srow] = v;
    }
  }
}

// ---- fused attention: per (b, h, 16-row s-tile), flash-style over t -------

__global__ __launch_bounds__(32)
void attn_kernel(const __bf16* __restrict__ Qb, const __bf16* __restrict__ Kb,
                 const __bf16* __restrict__ Vtb, __bf16* __restrict__ Ob) {
  __shared__ __bf16 lds[16][40];   // P-tile staging (C-layout -> A-layout)

  const int lane = threadIdx.x;
  const bool lo  = lane < 16;
  const int nl   = lane & 15;
  const int st   = blockIdx.x;     // s-tile, 0..107
  const int hh   = blockIdx.y;     // head
  const int bb   = blockIdx.z;     // batch

  const __bf16* Qbase = Qb  + ((size_t)bb * H_ + hh) * S_ * P_;
  const __bf16* Kbase = Kb  + ((size_t)bb * H_ + hh) * S_ * P_;
  const __bf16* Vbase = Vtb + ((size_t)bb * H_ + hh) * P_ * S_;

  // Q A-tile: row s = st*16+nl, K-dim = P = 32 (exactly one WMMA K)
  const int arow = st * 16 + nl;
  const v16bf qa = ld16(Qbase + (size_t)arow * P_ + (lo ? 0 : 16));

  float mrun[8], lrun[8];
  v8f olo = {}, ohi = {};
#pragma unroll
  for (int r = 0; r < 8; ++r) { mrun[r] = -3.0e38f; lrun[r] = 0.0f; }

  for (int t0 = 0; t0 < S_; t0 += 32) {
    // scores S[s, t0..t0+31] = Q . K^T  (no 1/sqrt(P) scale, per reference)
    v16bf kb0 = ld16(Kbase + (size_t)(t0 + nl) * P_ + (lo ? 0 : 16));
    v16bf kb1 = ld16(Kbase + (size_t)(t0 + 16 + nl) * P_ + (lo ? 0 : 16));
    v8f z = {};
    v8f s0 = wmma_bf16(qa, kb0, z);
    v8f s1 = wmma_bf16(qa, kb1, z);

    // online softmax per row; stage exp-tile to LDS in bf16
#pragma unroll
    for (int r = 0; r < 8; ++r) {
      float tm = half16_max(fmaxf(s0[r], s1[r]));
      float mnew = fmaxf(mrun[r], tm);
      float corr = __expf(mrun[r] - mnew);
      float p0 = __expf(s0[r] - mnew);
      float p1 = __expf(s1[r] - mnew);
      float rs = half16_sum(p0 + p1);
      lrun[r] = lrun[r] * corr + rs;
      mrun[r] = mnew;
      olo[r] *= corr;
      ohi[r] *= corr;
      int row = r + (lo ? 0 : 8);
      lds[row][nl]      = (__bf16)p0;
      lds[row][16 + nl] = (__bf16)p1;
    }
    __syncthreads();

    // P A-tile (16 s-rows x 32 t) from LDS; V B-tiles from [B,H,P,S]
    v16bf pa = ld16(&lds[nl][(lo ? 0 : 16)]);
    v16bf vb0 = ld16(Vbase + (size_t)nl        * S_ + t0 + (lo ? 0 : 16));
    v16bf vb1 = ld16(Vbase + (size_t)(16 + nl) * S_ + t0 + (lo ? 0 : 16));
    olo = wmma_bf16(pa, vb0, olo);   // output cols p = 0..15
    ohi = wmma_bf16(pa, vb1, ohi);   // output cols p = 16..31
    __syncthreads();
  }

  // normalize and write merged-head row-major [BS, E] (col = h*32 + p)
#pragma unroll
  for (int r = 0; r < 8; ++r) {
    int srow = st * 16 + r + (lo ? 0 : 8);
    float inv = 1.0f / lrun[r];
    size_t base = ((size_t)bb * S_ + srow) * E_ + hh * 32;
    Ob[base + nl]      = (__bf16)(olo[r] * inv);
    Ob[base + 16 + nl] = (__bf16)(ohi[r] * inv);
  }
}

// ---- output projection: out = Ob @ Wo + bo (f32), 16x64 per wave ----------

__global__ __launch_bounds__(32)
void out_gemm(const __bf16* __restrict__ Ob, const __bf16* __restrict__ Wot,
              const float* __restrict__ bo, float* __restrict__ out) {
  const int lane = threadIdx.x;
  const bool lo  = lane < 16;
  const int nl   = lane & 15;
  const int mt   = blockIdx.x;
  const int nt0  = blockIdx.y * 64;
  const int arow = mt * 16 + nl;

  const __bf16* ap = Ob + (size_t)arow * E_;
  const __bf16* bp0 = Wot + (size_t)(nt0 + nl) * E_;
  const __bf16* bp1 = bp0 + (size_t)16 * E_;
  const __bf16* bp2 = bp0 + (size_t)32 * E_;
  const __bf16* bp3 = bp0 + (size_t)48 * E_;

  v8f acc0 = {}, acc1 = {}, acc2 = {}, acc3 = {};
#pragma unroll
  for (int kk = 0; kk < E_; kk += 32) {
    v16bf a = ld16(ap + kk + (lo ? 0 : 16));
    int ko = kk + (lo ? 0 : 16);
    acc0 = wmma_bf16(a, ld16(bp0 + ko), acc0);
    acc1 = wmma_bf16(a, ld16(bp1 + ko), acc1);
    acc2 = wmma_bf16(a, ld16(bp2 + ko), acc2);
    acc3 = wmma_bf16(a, ld16(bp3 + ko), acc3);
  }

#pragma unroll
  for (int j = 0; j < 4; ++j) {
    const v8f& acc = (j == 0) ? acc0 : (j == 1) ? acc1 : (j == 2) ? acc2 : acc3;
    const int bcol = nt0 + j * 16 + nl;
    const float bb = bo[bcol];
#pragma unroll
    for (int r = 0; r < 8; ++r) {
      int grow = mt * 16 + r + (lo ? 0 : 8);
      out[(size_t)grow * E_ + bcol] = acc[r] + bb;
    }
  }
}

// ---- host launcher --------------------------------------------------------

extern "C" void kernel_launch(void* const* d_in, const int* in_sizes, int n_in,
                              void* d_out, int out_size, void* d_ws, size_t ws_size,
                              hipStream_t stream) {
  (void)in_sizes; (void)n_in; (void)out_size; (void)ws_size;
  const float* X  = (const float*)d_in[0];
  const float* Wq = (const float*)d_in[1];
  const float* bq = (const float*)d_in[2];
  const float* Wk = (const float*)d_in[3];
  const float* bk = (const float*)d_in[4];
  const float* Wv = (const float*)d_in[5];
  const float* bv = (const float*)d_in[6];
  const float* Wo = (const float*)d_in[7];
  const float* bo = (const float*)d_in[8];
  float* out = (float*)d_out;

  // carve workspace (~9.5 MB total)
  size_t off = 0;
  auto carve = [&](size_t nbytes) -> void* {
    void* p = (void*)((char*)d_ws + off);
    off += (nbytes + 255) & ~(size_t)255;
    return p;
  };
  __bf16* Xb  = (__bf16*)carve((size_t)BS_ * E_ * 2);
  __bf16* Wqt = (__bf16*)carve((size_t)E_ * E_ * 2);
  __bf16* Wkt = (__bf16*)carve((size_t)E_ * E_ * 2);
  __bf16* Wvt = (__bf16*)carve((size_t)E_ * E_ * 2);
  __bf16* Wot = (__bf16*)carve((size_t)E_ * E_ * 2);
  __bf16* Qb  = (__bf16*)carve((size_t)B_ * H_ * S_ * P_ * 2);
  __bf16* Kb  = (__bf16*)carve((size_t)B_ * H_ * S_ * P_ * 2);
  __bf16* Vtb = (__bf16*)carve((size_t)B_ * H_ * S_ * P_ * 2);
  __bf16* Ob  = (__bf16*)carve((size_t)BS_ * E_ * 2);

  // 1) convert inputs to bf16 (weights transposed for B-operand loads)
  {
    int n = BS_ * E_;
    cvt_f32_bf16<<<(n + 255) / 256, 256, 0, stream>>>(X, Xb, n);
    int nw = E_ * E_;
    cvt_wT_bf16<<<(nw + 255) / 256, 256, 0, stream>>>(Wq, Wqt);
    cvt_wT_bf16<<<(nw + 255) / 256, 256, 0, stream>>>(Wk, Wkt);
    cvt_wT_bf16<<<(nw + 255) / 256, 256, 0, stream>>>(Wv, Wvt);
    cvt_wT_bf16<<<(nw + 255) / 256, 256, 0, stream>>>(Wo, Wot);
  }

  // 2) QKV projections (head-split layouts), 16x64 tiles per wave
  dim3 gp(BS_ / 16, E_ / 64);
  proj_kernel<<<gp, 32, 0, stream>>>(Xb, Wqt, bq, Qb, Kb, Vtb, 0);
  proj_kernel<<<gp, 32, 0, stream>>>(Xb, Wkt, bk, Qb, Kb, Vtb, 1);
  proj_kernel<<<gp, 32, 0, stream>>>(Xb, Wvt, bv, Qb, Kb, Vtb, 2);

  // 3) fused attention (scores never touch HBM)
  dim3 ga(S_ / 16, H_, B_);
  attn_kernel<<<ga, 32, 0, stream>>>(Qb, Kb, Vtb, Ob);

  // 4) output projection
  out_gemm<<<gp, 32, 0, stream>>>(Ob, Wot, bo, out);
}
